// SDOT_77704548319292
// MI455X (gfx1250) — matrix operand order repaired and checked by reading, not compiled, not run on hardware
//
#include <hip/hip_runtime.h>
#include <math.h>

#define N_SRC 16384
#define M_TGT 8192
#define DIM   128
#define REGP  0.05f
#define ROWS_PER_BLOCK 32

typedef __attribute__((ext_vector_type(8)))  short  v8s;
typedef __attribute__((ext_vector_type(16))) short  v16s;
typedef __attribute__((ext_vector_type(16))) __bf16 v16bf;
typedef __attribute__((ext_vector_type(8)))  float  v8f;

// Round-to-nearest-even fp32 -> bf16 (as raw ushort)
static __device__ __forceinline__ unsigned short f2bf(float x) {
    unsigned u = __builtin_bit_cast(unsigned, x);
    u = (u + 0x7FFFu + ((u >> 16) & 1u)) >> 16;
    return (unsigned short)u;
}
static __device__ __forceinline__ float bf2f(unsigned short b) {
    return __builtin_bit_cast(float, (unsigned)b << 16);
}

// Build a 16-element bf16 fragment from two contiguous 8-element (16B) runs.
static __device__ __forceinline__ v16bf ld_frag(const unsigned short* p0,
                                                const unsigned short* p1) {
    const v8s lo = *(const v8s*)p0;        // global_load_b128
    const v8s hi = *(const v8s*)p1;        // global_load_b128
    const v16s w = __builtin_shufflevector(lo, hi,
        0, 1, 2, 3, 4, 5, 6, 7, 8, 9, 10, 11, 12, 13, 14, 15);
    return __builtin_bit_cast(v16bf, w);
}

static __device__ __forceinline__ v8f wmma_bf16(v16bf a, v16bf b, v8f c) {
    return __builtin_amdgcn_wmma_f32_16x16x32_bf16(
        false, a, false, b, (short)0, c, false, false);
}

// ---------------------------------------------------------------------------
// Kernel A: per-row squared norms (fp32 inputs, full precision).
// ---------------------------------------------------------------------------
__global__ __launch_bounds__(256)
void norms_kernel(const float* __restrict__ src, const float* __restrict__ tgt,
                  float* __restrict__ sx2, float* __restrict__ ty2) {
    const int wave = threadIdx.x >> 5;
    const int lane = threadIdx.x & 31;
    const int row  = blockIdx.x * 8 + wave;

    const float* p;
    float* outp;
    int idx;
    if (row < N_SRC) {
        p = src + (size_t)row * DIM; outp = sx2; idx = row;
    } else {
        const int m = row - N_SRC;
        if (m >= M_TGT) return;
        p = tgt + (size_t)m * DIM; outp = ty2; idx = m;
    }
    const float4 v = ((const float4*)p)[lane];
    float s = v.x * v.x + v.y * v.y + v.z * v.z + v.w * v.w;
#pragma unroll
    for (int d = 16; d >= 1; d >>= 1) s += __shfl_xor(s, d, 32);
    if (lane == 0) outp[idx] = s;
}

// ---------------------------------------------------------------------------
// Kernel A2: split fp32 -> bf16 (hi) + bf16 (residual lo) for src and tgt.
// ---------------------------------------------------------------------------
__global__ __launch_bounds__(256)
void convert_kernel(const float* __restrict__ src, const float* __restrict__ tgt,
                    unsigned short* __restrict__ sH, unsigned short* __restrict__ sL,
                    unsigned short* __restrict__ tH, unsigned short* __restrict__ tL) {
    const size_t t = (size_t)blockIdx.x * 256 + threadIdx.x;
    const size_t i = t * 4;
    const size_t nsrc = (size_t)N_SRC * DIM;

    const float* p;
    unsigned short *h, *l;
    size_t off;
    if (i < nsrc) {
        p = src; h = sH; l = sL; off = i;
    } else {
        off = i - nsrc;
        if (off >= (size_t)M_TGT * DIM) return;
        p = tgt; h = tH; l = tL;
    }
    const float4 v = *(const float4*)(p + off);
    const unsigned short h0 = f2bf(v.x), h1 = f2bf(v.y), h2 = f2bf(v.z), h3 = f2bf(v.w);
    const unsigned short l0 = f2bf(v.x - bf2f(h0));
    const unsigned short l1 = f2bf(v.y - bf2f(h1));
    const unsigned short l2 = f2bf(v.z - bf2f(h2));
    const unsigned short l3 = f2bf(v.w - bf2f(h3));
    *(ushort4*)(h + off) = make_ushort4(h0, h1, h2, h3);
    *(ushort4*)(l + off) = make_ushort4(l0, l1, l2, l3);
}

// ---------------------------------------------------------------------------
// Kernel B: scale = mean(ty2); scalars = {1/scale, 2/(scale*REG)};
// adj[m] = (psi[m] - ty2[m]/scale) / REG.
// ---------------------------------------------------------------------------
__global__ __launch_bounds__(256)
void scale_adj_kernel(const float* __restrict__ psi, const float* __restrict__ ty2,
                      float* __restrict__ adj, float* __restrict__ scalars) {
    __shared__ float red[256];
    __shared__ float sc[2];

    float s = 0.f;
    for (int m = threadIdx.x; m < M_TGT; m += 256) s += ty2[m];
    red[threadIdx.x] = s;
    __syncthreads();
    for (int off = 128; off >= 1; off >>= 1) {
        if (threadIdx.x < (unsigned)off) red[threadIdx.x] += red[threadIdx.x + off];
        __syncthreads();
    }
    if (threadIdx.x == 0) {
        const float scale = red[0] / (float)M_TGT;
        sc[0] = 1.0f / scale;
        sc[1] = 2.0f / (scale * REGP);
        scalars[0] = sc[0];
        scalars[1] = sc[1];
    }
    __syncthreads();
    const float inv_scale = sc[0];
    const float inv_reg   = 1.0f / REGP;
    for (int m = threadIdx.x; m < M_TGT; m += 256)
        adj[m] = (psi[m] - ty2[m] * inv_scale) * inv_reg;
}

// ---------------------------------------------------------------------------
// Kernel C: fused GEMM (V_WMMA_F32_16X16X32_BF16, bf16x3 split) + online LSE.
// Block = 32 source rows (two 16-row A fragment sets resident in registers,
// two accumulators sharing every B fragment -> half the L2 traffic, two
// independent WMMA accumulate chains). 8 waves sweep disjoint 16-col tiles.
// ---------------------------------------------------------------------------
__global__ __launch_bounds__(256)
void lse_gemm_kernel(const unsigned short* __restrict__ srcH,
                     const unsigned short* __restrict__ srcL,
                     const unsigned short* __restrict__ tgtH,
                     const unsigned short* __restrict__ tgtL,
                     const float* __restrict__ adj, const float* __restrict__ sx2,
                     const float* __restrict__ scalars, float* __restrict__ val) {
    __shared__ float redm[8 * ROWS_PER_BLOCK];
    __shared__ float reds[8 * ROWS_PER_BLOCK];

    const int tid  = threadIdx.x;
    const int wave = tid >> 5;
    const int lane = tid & 31;
    const int half = lane >> 4;        // 0: lanes 0-15, 1: lanes 16-31
    const int l16  = lane & 15;
    const int row0 = blockIdx.x * ROWS_PER_BLOCK;

    // A fragments (16-bit A 16x32 layout): row M = l16 (+16 for second set);
    // K chunks at 8h..8h+7 and 16+8h..16+8h+7 within each 32-K block.
    const unsigned short* aH0p = srcH + (size_t)(row0 + l16) * DIM;
    const unsigned short* aL0p = srcL + (size_t)(row0 + l16) * DIM;
    const unsigned short* aH1p = srcH + (size_t)(row0 + 16 + l16) * DIM;
    const unsigned short* aL1p = srcL + (size_t)(row0 + 16 + l16) * DIM;
    v16bf aH0[4], aL0[4], aH1[4], aL1[4];
#pragma unroll
    for (int c = 0; c < 4; ++c) {
        const int k0 = 32 * c + 8 * half;
        aH0[c] = ld_frag(aH0p + k0, aH0p + k0 + 16);
        aL0[c] = ld_frag(aL0p + k0, aL0p + k0 + 16);
        aH1[c] = ld_frag(aH1p + k0, aH1p + k0 + 16);
        aL1[c] = ld_frag(aL1p + k0, aL1p + k0 + 16);
    }

    const float beta = scalars[1];

    float mx0[8], sm0[8], mx1[8], sm1[8];
#pragma unroll
    for (int r = 0; r < 8; ++r) {
        mx0[r] = -3.0e38f; sm0[r] = 0.f;
        mx1[r] = -3.0e38f; sm1[r] = 0.f;
    }

    for (int j = 0; j < M_TGT / 128; ++j) {
        const int col = wave * 16 + j * 128 + l16;   // this lane's target column
        const float adjv = adj[col];
        const unsigned short* bHrow = tgtH + (size_t)col * DIM;
        const unsigned short* bLrow = tgtL + (size_t)col * DIM;

        if (j + 1 < M_TGT / 128) {
            __builtin_prefetch(bHrow + 128 * DIM, 0, 3);
            __builtin_prefetch(bLrow + 128 * DIM, 0, 3);
        }

        v8f acc0 = {};
        v8f acc1 = {};
#pragma unroll
        for (int c = 0; c < 4; ++c) {
            // B fragments (B 32x16): column n = l16, K chunks at
            // 8h..8h+7 and 16+8h..16+8h+7 within each 32-K block.
            const int k0 = 32 * c + 8 * half;
            const v16bf bh = ld_frag(bHrow + k0, bHrow + k0 + 16);
            const v16bf bl = ld_frag(bLrow + k0, bLrow + k0 + 16);
            // bf16x3: Ah*Bh + Al*Bh + Ah*Bl; interleave the two acc chains.
            acc0 = wmma_bf16(aH0[c], bh, acc0);
            acc1 = wmma_bf16(aH1[c], bh, acc1);
            acc0 = wmma_bf16(aL0[c], bh, acc0);
            acc1 = wmma_bf16(aL1[c], bh, acc1);
            acc0 = wmma_bf16(aH0[c], bl, acc0);
            acc1 = wmma_bf16(aH1[c], bl, acc1);
        }

        // Online logsumexp (1 transcendental per slot via |d| trick).
#pragma unroll
        for (int r = 0; r < 8; ++r) {
            {
                const float t = fmaf(beta, acc0[r], adjv);
                const float d = t - mx0[r];
                const float e = __expf(0.f - fabsf(d));
                const bool  g = d > 0.f;
                sm0[r] = g ? fmaf(sm0[r], e, 1.f) : (sm0[r] + e);
                mx0[r] = g ? t : mx0[r];
            }
            {
                const float t = fmaf(beta, acc1[r], adjv);
                const float d = t - mx1[r];
                const float e = __expf(0.f - fabsf(d));
                const bool  g = d > 0.f;
                sm1[r] = g ? fmaf(sm1[r], e, 1.f) : (sm1[r] + e);
                mx1[r] = g ? t : mx1[r];
            }
        }
    }

    // Merge across the 16 lanes of each half (columns of the tile).
#pragma unroll
    for (int set = 0; set < 2; ++set) {
#pragma unroll
        for (int r = 0; r < 8; ++r) {
            float m = set ? mx1[r] : mx0[r];
            float s = set ? sm1[r] : sm0[r];
#pragma unroll
            for (int d = 1; d < 16; d <<= 1) {
                const float om = __shfl_xor(m, d, 32);
                const float os = __shfl_xor(s, d, 32);
                const float nm = fmaxf(m, om);
                s = s * __expf(m - nm) + os * __expf(om - nm);
                m = nm;
            }
            if (l16 == 0) {
                const int srow = set * 16 + r + 8 * half;
                redm[wave * ROWS_PER_BLOCK + srow] = m;
                reds[wave * ROWS_PER_BLOCK + srow] = s;
            }
        }
    }
    __syncthreads();

    // Merge the 8 waves' column partitions; finalize the 32 rows.
    if (tid < ROWS_PER_BLOCK) {
        float m = redm[tid], s = reds[tid];
#pragma unroll
        for (int w = 1; w < 8; ++w) {
            const float om = redm[w * ROWS_PER_BLOCK + tid];
            const float os = reds[w * ROWS_PER_BLOCK + tid];
            const float nm = fmaxf(m, om);
            s = s * __expf(m - nm) + os * __expf(om - nm);
            m = nm;
        }
        const float inv_scale = scalars[0];
        const float lse = __logf(s) + m;
        // -REG * (lse - sx2/(scale*REG)) = -REG*lse + sx2/scale
        val[row0 + tid] = -REGP * lse + sx2[row0 + tid] * inv_scale;
    }
}

// ---------------------------------------------------------------------------
// Kernel D: scalar finalize: mean(val) + mean(psi) + REG*log(M).
// ---------------------------------------------------------------------------
__global__ __launch_bounds__(256)
void finalize_kernel(const float* __restrict__ val, const float* __restrict__ psi,
                     float* __restrict__ out) {
    __shared__ float red[256];
    float sv = 0.f, sp = 0.f;
    for (int i = threadIdx.x; i < N_SRC; i += 256) sv += val[i];
    for (int i = threadIdx.x; i < M_TGT; i += 256) sp += psi[i];

    red[threadIdx.x] = sv;
    __syncthreads();
    for (int off = 128; off >= 1; off >>= 1) {
        if (threadIdx.x < (unsigned)off) red[threadIdx.x] += red[threadIdx.x + off];
        __syncthreads();
    }
    const float total_v = red[0];
    __syncthreads();

    red[threadIdx.x] = sp;
    __syncthreads();
    for (int off = 128; off >= 1; off >>= 1) {
        if (threadIdx.x < (unsigned)off) red[threadIdx.x] += red[threadIdx.x + off];
        __syncthreads();
    }
    if (threadIdx.x == 0) {
        const float mean_v = total_v / (float)N_SRC;
        const float mean_p = red[0] / (float)M_TGT;
        out[0] = mean_v + mean_p + REGP * __logf((float)M_TGT);
    }
}

// ---------------------------------------------------------------------------
extern "C" void kernel_launch(void* const* d_in, const int* in_sizes, int n_in,
                              void* d_out, int out_size, void* d_ws, size_t ws_size,
                              hipStream_t stream) {
    const float* src = (const float*)d_in[0];   // [N, D]
    const float* tgt = (const float*)d_in[1];   // [M, D]
    const float* psi = (const float*)d_in[2];   // [M]
    float* out = (float*)d_out;

    // fp32 workspace
    float* w       = (float*)d_ws;
    float* sx2     = w;                   // N
    float* ty2     = sx2 + N_SRC;         // M
    float* adj     = ty2 + M_TGT;         // M
    float* valv    = adj + M_TGT;         // N
    float* scalars = valv + N_SRC;        // 32 (padded)

    // bf16 workspace (16B aligned)
    unsigned short* sH = (unsigned short*)(w + 49184);
    unsigned short* sL = sH + (size_t)N_SRC * DIM;
    unsigned short* tH = sL + (size_t)N_SRC * DIM;
    unsigned short* tL = tH + (size_t)M_TGT * DIM;

    norms_kernel<<<(N_SRC + M_TGT) / 8, 256, 0, stream>>>(src, tgt, sx2, ty2);
    convert_kernel<<<(N_SRC + M_TGT) * (DIM / 4) / 256, 256, 0, stream>>>(
        src, tgt, sH, sL, tH, tL);
    scale_adj_kernel<<<1, 256, 0, stream>>>(psi, ty2, adj, scalars);
    lse_gemm_kernel<<<N_SRC / ROWS_PER_BLOCK, 256, 0, stream>>>(
        sH, sL, tH, tL, adj, sx2, scalars, valv);
    finalize_kernel<<<1, 256, 0, stream>>>(valv, psi, out);
}
